// DSTFT_33097017983244
// MI455X (gfx1250) — matrix operand order repaired and checked by reading, not compiled, not run on hardware
//
#include <hip/hip_runtime.h>
#include <hip/hip_bf16.h>

typedef __attribute__((ext_vector_type(16))) _Float16 v16h;
typedef __attribute__((ext_vector_type(8)))  float    v8f;

#define DS_N_FFT   1024
#define DS_FREQ    513
#define DS_FRAMES  256
#define DS_BATCH   4
#define DS_SIGLEN  (256 * 256 + 1024)   /* 66560 */
#define DS_FTILES  ((DS_FREQ + 15) / 16) /* 33 */

__global__ __launch_bounds__(32)
void dstft_wmma_kernel(const float* __restrict__ x,
                       const float* __restrict__ raw_win,
                       const float* __restrict__ raw_hop,
                       float* __restrict__ out)
{
    const int ftile = blockIdx.x;   // 0..32  (16 freqs per tile)
    const int t     = blockIdx.y;   // 0..255 (frame index)
    const int lane  = threadIdx.x;  // 0..31  (wave32)

    const float TWO_PI = 6.28318530717958647692f;

    // ---- scalar frame parameters (uniform across wave) ----
    const float hop  = 1.0f + (1.0f / (1.0f + __expf(-raw_hop[0]))) * (256.0f - 1.0f);
    float pos = (float)t * hop;
    pos = fminf(fmaxf(pos, 0.0f), (float)(DS_SIGLEN - DS_N_FFT));
    const int   idx0 = (int)floorf(pos);
    const float frac = pos - (float)idx0;

    const int halfsel = lane >> 4;   // 0: lanes 0-15, 1: lanes 16-31
    const int l15     = lane & 15;

    // ---- B-matrix role: this lane's frequency bin ----
    const int   f      = ftile * 16 + l15;
    const int   fc     = (f < DS_FREQ) ? f : (DS_FREQ - 1);
    const float fvalid = (f < DS_FREQ) ? 1.0f : 0.0f;
    const float theta      = 10.0f + (1.0f / (1.0f + __expf(-raw_win[fc]))) * (1024.0f - 10.0f);
    const float inv_theta  = __frcp_rn(theta);
    const float half_theta = 0.5f * theta;
    const float ang_coef   = (float)fc * (TWO_PI / (float)DS_N_FFT); // ang = ang_coef * n

    // ---- A-matrix role: this lane's batch row ----
    // Rows 4..15 of D are never stored, so unused lanes just duplicate row 0:
    // loads are UNCONDITIONAL -> no EXEC masking, loads pipeline freely.
    const int mrow = (l15 < DS_BATCH) ? l15 : 0;
    const float* xrow = x + (size_t)mrow * DS_SIGLEN + idx0;

    const float center = (float)(DS_N_FFT - 1) * 0.5f; // 511.5

    v8f acc_re = {0.f,0.f,0.f,0.f,0.f,0.f,0.f,0.f};
    v8f acc_im = {0.f,0.f,0.f,0.f,0.f,0.f,0.f,0.f};

    const int akb = halfsel ? 8  : 0;  // A K-base (16-bit A 16x32 layout)
    const int bkb = halfsel ? 8  : 0;  // B K-base (16-bit B 32x16 layout)

#pragma unroll 2
    for (int n0 = 0; n0 < DS_N_FFT; n0 += 32) {
        // Unconditional prefetch of the next chunk (stays inside the row:
        // idx0 <= 65536, worst offset 992+64+23 < row length 66560).
        __builtin_prefetch(xrow + n0 + 64, 0, 0);

        // ---- A fragment: frames[b, t, n0+K] (16-bit A 16x32 layout) ----
        v16h a;
#pragma unroll
        for (int e = 0; e < 16; ++e) {
            const int K = akb + (e < 8 ? e : e + 8);
            a[e] = (_Float16)xrow[n0 + K];
        }

        // ---- B fragments: w(f,t,n)*cos(ang), w(f,t,n)*sin(ang) ----
        v16h bc, bs;
#pragma unroll
        for (int e = 0; e < 16; ++e) {
            const int   n  = n0 + bkb + (e < 8 ? e : e + 8);
            const float nf = (float)n;
            const float d  = nf - center - frac;
            const float wr = 0.5f + 0.5f * __cosf(TWO_PI * d * inv_theta);
            float w  = (fabsf(d) <= half_theta) ? wr : 0.0f;   // branchless
            w *= fvalid;
            const float ang = ang_coef * nf;
            const float c = __cosf(ang);
            const float s = __sinf(ang);
            bc[e] = (_Float16)(w * c);
            bs[e] = (_Float16)(w * s);
        }

        // D = A x B + C  (16x16x32 f16 -> f32 accum), EXEC all ones here
        acc_re = __builtin_amdgcn_wmma_f32_16x16x32_f16(
                     false, a, false, bc, (short)0, acc_re, false, false);
        acc_im = __builtin_amdgcn_wmma_f32_16x16x32_f16(
                     false, a, false, bs, (short)0, acc_im, false, false);
    }

    // ---- store: C/D f32 layout -> lanes 0-15 hold rows M=0..7 in v[0..7];
    // only rows 0..3 (batch) are meaningful, column N = lane = freq in tile.
    if (halfsel == 0 && f < DS_FREQ) {
        const size_t spec_n = (size_t)DS_BATCH * DS_FREQ * DS_FRAMES; // 525312
#pragma unroll
        for (int b = 0; b < DS_BATCH; ++b) {
            const float re = acc_re[b];
            const float im = -acc_im[b];                // im = -sum(...)
            const size_t idx = ((size_t)b * DS_FREQ + f) * DS_FRAMES + t;
            out[idx] = sqrtf(re * re + im * im) + 1e-12f;    // spec
            out[spec_n + 2 * idx]     = re;                  // stft (complex64)
            out[spec_n + 2 * idx + 1] = im;
        }
    }
}

extern "C" void kernel_launch(void* const* d_in, const int* in_sizes, int n_in,
                              void* d_out, int out_size, void* d_ws, size_t ws_size,
                              hipStream_t stream) {
    (void)in_sizes; (void)n_in; (void)d_ws; (void)ws_size; (void)out_size;
    const float* x        = (const float*)d_in[0];
    const float* raw_win  = (const float*)d_in[1];
    const float* raw_hop  = (const float*)d_in[2];
    float*       out      = (float*)d_out;

    dim3 grid(DS_FTILES, DS_FRAMES);  // 33 x 256 workgroups
    dim3 block(32);                   // one wave32 per workgroup
    dstft_wmma_kernel<<<grid, block, 0, stream>>>(x, raw_win, raw_hop, out);
}
